// TransformerEncoderLayer_11055245819986
// MI455X (gfx1250) — matrix operand order repaired and checked by reading, not compiled, not run on hardware
//
#include <hip/hip_runtime.h>

// ---------------------------------------------------------------------------
// Transformer encoder layer for gfx1250 (MI455X), bf16 WMMA everywhere.
// B=2, S=2048, H=768, NH=12, D=64, FF=3072.
// ---------------------------------------------------------------------------

typedef unsigned short u16;
typedef __attribute__((ext_vector_type(16))) __bf16 v16bf;
typedef __attribute__((ext_vector_type(8)))  __bf16 v8bf;
typedef __attribute__((ext_vector_type(8)))  float  v8f;

#define WMMA_BF16(A_, B_, C_) \
  __builtin_amdgcn_wmma_f32_16x16x32_bf16(false, (A_), false, (B_), (short)0, (C_), false, false)

union AFrag { v16bf v; v8bf h[2]; };

__device__ __forceinline__ u16 f2bf(float f) {
  union { float f; unsigned u; } x; x.f = f;
  unsigned r = x.u + 0x7FFFu + ((x.u >> 16) & 1u);
  return (u16)(r >> 16);
}

// --------------------------- elementwise converts ---------------------------

__global__ void cvt_bf16_kernel(const float* __restrict__ in, u16* __restrict__ o, long n) {
  long i = (long)blockIdx.x * blockDim.x + threadIdx.x;
  if (i < n) o[i] = f2bf(in[i]);
}

// out[z][c][r] = bf16(in[z][r][c])  (per-z matrix transpose + convert)
__global__ void transpose_bf16_kernel(const float* __restrict__ in, u16* __restrict__ o,
                                      int rows, int cols) {
  long total = (long)rows * cols;
  const float* iz = in + (long)blockIdx.z * total;
  u16* oz = o + (long)blockIdx.z * total;
  long i = (long)blockIdx.x * blockDim.x + threadIdx.x;
  if (i < total) {
    int c = (int)(i / rows), r = (int)(i % rows);
    oz[i] = f2bf(iz[(long)r * cols + c]);
  }
}

// ------------------------------- bf16 GEMM ---------------------------------
// C[M,N] = A[M,K] (bf16 rm) @ Bt[N,K]^T (bf16, pre-transposed) + bias
// 4 waves/block, 64x64 block tile, 32x32 per wave, K-step 32.
// vtmode=1: bf16 store transposed per-batch for V: [(m>>11)*64+n][2048] + (m&2047)
__global__ __launch_bounds__(128)
void gemm_bf16_wmma(const u16* __restrict__ A, const u16* __restrict__ Bt,
                    const float* __restrict__ bias,
                    float* __restrict__ Cf, u16* __restrict__ Cbf,
                    int M, int N, int K, int relu, int vtmode,
                    long strideBt, long strideF, long strideBf) {
  Bt += (long)blockIdx.z * strideBt;
  if (Cf)  Cf  += (long)blockIdx.z * strideF;
  if (Cbf) Cbf += (long)blockIdx.z * strideBf;

  const int lane = threadIdx.x & 31;
  const int wv   = threadIdx.x >> 5;
  const int mBase = blockIdx.y * 64 + (wv >> 1) * 32;
  const int nBase = blockIdx.x * 64 + (wv & 1) * 32;
  const int nl = lane & 15;
  const int hf = lane >> 4;
  const int b0 = hf * 8;

  v8f acc[2][2];
#pragma unroll
  for (int i = 0; i < 2; ++i)
#pragma unroll
    for (int j = 0; j < 2; ++j)
#pragma unroll
      for (int r = 0; r < 8; ++r) acc[i][j][r] = 0.0f;

  const u16* Arow0 = A  + (long)(mBase + nl)      * K;
  const u16* Arow1 = A  + (long)(mBase + 16 + nl) * K;
  const u16* Brow0 = Bt + (long)(nBase + nl)      * K;
  const u16* Brow1 = Bt + (long)(nBase + 16 + nl) * K;

  for (int kb = 0; kb < K; kb += 32) {
    AFrag a0, a1;
    a0.h[0] = *(const v8bf*)(Arow0 + kb + 2 * b0);
    a0.h[1] = *(const v8bf*)(Arow0 + kb + 8 + 2 * b0);
    a1.h[0] = *(const v8bf*)(Arow1 + kb + 2 * b0);
    a1.h[1] = *(const v8bf*)(Arow1 + kb + 8 + 2 * b0);
    const int koff = kb + hf * 16;
    v16bf bf0 = *(const v16bf*)(Brow0 + koff);
    v16bf bf1 = *(const v16bf*)(Brow1 + koff);
    acc[0][0] = WMMA_BF16(a0.v, bf0, acc[0][0]);
    acc[0][1] = WMMA_BF16(a0.v, bf1, acc[0][1]);
    acc[1][0] = WMMA_BF16(a1.v, bf0, acc[1][0]);
    acc[1][1] = WMMA_BF16(a1.v, bf1, acc[1][1]);
  }

#pragma unroll
  for (int mi = 0; mi < 2; ++mi)
#pragma unroll
    for (int ni = 0; ni < 2; ++ni) {
      const int n = nBase + ni * 16 + nl;
      const float bb = bias ? bias[n] : 0.0f;
#pragma unroll
      for (int r = 0; r < 8; ++r) {
        const int m = mBase + mi * 16 + hf * 8 + r;
        float v = acc[mi][ni][r] + bb;
        if (relu) v = fmaxf(v, 0.0f);
        if (Cf) Cf[(long)m * N + n] = v;
        if (Cbf) {
          if (vtmode) Cbf[((long)(m >> 11) * 64 + n) * 2048 + (m & 2047)] = f2bf(v);
          else        Cbf[(long)m * N + n] = f2bf(v);
        }
      }
    }
}

// ---------------------------- flash attention ------------------------------
// One wave per (b, head, 16-query block). Q/K: [head][4096][64] bf16,
// Vt: [head*2+b][64][2048] bf16. Output: attn concat bf16 [4096][768].
__global__ __launch_bounds__(32)
void flash_attn_kernel(const u16* __restrict__ Q, const u16* __restrict__ K,
                       const u16* __restrict__ Vt, u16* __restrict__ outbf) {
  const int lane = threadIdx.x & 31;
  const int nl = lane & 15;
  const int hf = lane >> 4;
  const int b0 = hf * 8;
  const int head = blockIdx.y;
  const int b    = blockIdx.z;
  const int q0   = blockIdx.x * 16;

  const u16* Qb = Q  + ((long)head * 4096 + (long)b * 2048) * 64;
  const u16* Kb = K  + ((long)head * 4096 + (long)b * 2048) * 64;
  const u16* Vb = Vt + ((long)(head * 2 + b) * 64) * 2048;

  // Q A-fragments (16 rows x 64 dims -> two K=32 fragments), loaded once.
  const long qr = (long)(q0 + nl) * 64;
  AFrag qa0, qa1;
  qa0.h[0] = *(const v8bf*)(Qb + qr + 2 * b0);
  qa0.h[1] = *(const v8bf*)(Qb + qr + 8 + 2 * b0);
  qa1.h[0] = *(const v8bf*)(Qb + qr + 32 + 2 * b0);
  qa1.h[1] = *(const v8bf*)(Qb + qr + 40 + 2 * b0);

  v8f o[4];
#pragma unroll
  for (int dt = 0; dt < 4; ++dt)
#pragma unroll
    for (int r = 0; r < 8; ++r) o[dt][r] = 0.0f;

  float mrow[8], lrow[8];
#pragma unroll
  for (int r = 0; r < 8; ++r) { mrow[r] = -3.0e38f; lrow[r] = 0.0f; }

  __shared__ __align__(16) u16 Pl[16 * 32];

  for (int kb0 = 0; kb0 < 2048; kb0 += 32) {
    // scores for 32 keys as two 16x16 tiles
    v8f s0, s1;
#pragma unroll
    for (int r = 0; r < 8; ++r) { s0[r] = 0.0f; s1[r] = 0.0f; }
    {
      const long kr0 = (long)(kb0 + nl) * 64;
      const long kr1 = (long)(kb0 + 16 + nl) * 64;
      const int d0 = hf * 16;
      v16bf k00 = *(const v16bf*)(Kb + kr0 + d0);
      v16bf k01 = *(const v16bf*)(Kb + kr0 + 32 + d0);
      v16bf k10 = *(const v16bf*)(Kb + kr1 + d0);
      v16bf k11 = *(const v16bf*)(Kb + kr1 + 32 + d0);
      s0 = WMMA_BF16(qa0.v, k00, s0);
      s0 = WMMA_BF16(qa1.v, k01, s0);
      s1 = WMMA_BF16(qa0.v, k10, s1);
      s1 = WMMA_BF16(qa1.v, k11, s1);
    }
    // streaming softmax per row (rows hf*8 .. hf*8+7 live in this half-wave)
    float p0[8], p1[8], osc[8];
#pragma unroll
    for (int r = 0; r < 8; ++r) {
      float a0v = s0[r] * 0.125f;
      float a1v = s1[r] * 0.125f;
      float mx = fmaxf(a0v, a1v);
      mx = fmaxf(mx, __shfl_xor(mx, 1, 32));
      mx = fmaxf(mx, __shfl_xor(mx, 2, 32));
      mx = fmaxf(mx, __shfl_xor(mx, 4, 32));
      mx = fmaxf(mx, __shfl_xor(mx, 8, 32));
      float mn = fmaxf(mrow[r], mx);
      float sc = __expf(mrow[r] - mn);
      p0[r] = __expf(a0v - mn);
      p1[r] = __expf(a1v - mn);
      float sum = p0[r] + p1[r];
      sum += __shfl_xor(sum, 1, 32);
      sum += __shfl_xor(sum, 2, 32);
      sum += __shfl_xor(sum, 4, 32);
      sum += __shfl_xor(sum, 8, 32);
      lrow[r] = lrow[r] * sc + sum;
      mrow[r] = mn;
      osc[r] = sc;
    }
#pragma unroll
    for (int dt = 0; dt < 4; ++dt)
#pragma unroll
      for (int r = 0; r < 8; ++r) o[dt][r] *= osc[r];

    // P (C layout) -> LDS -> A-fragment layout
    __syncthreads();
#pragma unroll
    for (int r = 0; r < 8; ++r) {
      const int m = r + hf * 8;
      Pl[m * 32 + nl]      = f2bf(p0[r]);
      Pl[m * 32 + 16 + nl] = f2bf(p1[r]);
    }
    __syncthreads();
    AFrag pa;
    pa.h[0] = *(const v8bf*)(Pl + nl * 32 + 2 * b0);
    pa.h[1] = *(const v8bf*)(Pl + nl * 32 + 8 + 2 * b0);

    const int kd = kb0 + hf * 16;
#pragma unroll
    for (int dt = 0; dt < 4; ++dt) {
      v16bf vf = *(const v16bf*)(Vb + (long)(dt * 16 + nl) * 2048 + kd);
      o[dt] = WMMA_BF16(pa.v, vf, o[dt]);
    }
  }

  // normalize and write concat output (bf16)
#pragma unroll
  for (int r = 0; r < 8; ++r) {
    const float inv = 1.0f / lrow[r];
    const long row = (long)b * 2048 + q0 + hf * 8 + r;
    u16* orow = outbf + row * 768 + head * 64;
#pragma unroll
    for (int dt = 0; dt < 4; ++dt)
      orow[dt * 16 + nl] = f2bf(o[dt][r] * inv);
  }
}

// ---------------------- fused residual add + layernorm ---------------------
// One 256-thread block per row of 768. Emits f32 and/or bf16.
__global__ __launch_bounds__(256)
void add_ln_kernel(const float* __restrict__ a, const float* __restrict__ res,
                   const float* __restrict__ g, const float* __restrict__ bta,
                   float* __restrict__ outf, u16* __restrict__ outbf) {
  const int row = blockIdx.x;
  const int tid = threadIdx.x;
  const float* ar = a + (long)row * 768;
  const float* rr = res + (long)row * 768;
  float v0 = ar[tid] + rr[tid];
  float v1 = ar[tid + 256] + rr[tid + 256];
  float v2 = ar[tid + 512] + rr[tid + 512];
  __shared__ float red[256];
  red[tid] = v0 + v1 + v2;
  __syncthreads();
  for (int off = 128; off > 0; off >>= 1) {
    if (tid < off) red[tid] += red[tid + off];
    __syncthreads();
  }
  const float mu = red[0] * (1.0f / 768.0f);
  __syncthreads();
  const float d0 = v0 - mu, d1 = v1 - mu, d2 = v2 - mu;
  red[tid] = d0 * d0 + d1 * d1 + d2 * d2;
  __syncthreads();
  for (int off = 128; off > 0; off >>= 1) {
    if (tid < off) red[tid] += red[tid + off];
    __syncthreads();
  }
  const float rstd = rsqrtf(red[0] * (1.0f / 768.0f) + 1e-5f);
  const long base = (long)row * 768;
  float o;
  o = d0 * rstd * g[tid] + bta[tid];
  if (outf) outf[base + tid] = o;
  if (outbf) outbf[base + tid] = f2bf(o);
  o = d1 * rstd * g[tid + 256] + bta[tid + 256];
  if (outf) outf[base + tid + 256] = o;
  if (outbf) outbf[base + tid + 256] = f2bf(o);
  o = d2 * rstd * g[tid + 512] + bta[tid + 512];
  if (outf) outf[base + tid + 512] = o;
  if (outbf) outbf[base + tid + 512] = f2bf(o);
}

// --------------------------------- launch ----------------------------------

extern "C" void kernel_launch(void* const* d_in, const int* in_sizes, int n_in,
                              void* d_out, int out_size, void* d_ws, size_t ws_size,
                              hipStream_t stream) {
  constexpr int kBS = 4096;   // B*S
  constexpr int kH  = 768;
  constexpr int kNH = 12;
  constexpr int kD  = 64;
  constexpr int kFF = 3072;

  const float* x   = (const float*)d_in[0];
  const float* Wq  = (const float*)d_in[1];
  const float* Wk  = (const float*)d_in[2];
  const float* Wv  = (const float*)d_in[3];
  const float* Wo  = (const float*)d_in[4];
  const float* bo  = (const float*)d_in[5];
  const float* g1  = (const float*)d_in[6];
  const float* be1 = (const float*)d_in[7];
  const float* W1  = (const float*)d_in[8];
  const float* b1  = (const float*)d_in[9];
  const float* W2  = (const float*)d_in[10];
  const float* b2  = (const float*)d_in[11];
  const float* g2  = (const float*)d_in[12];
  const float* be2 = (const float*)d_in[13];
  float* outp = (float*)d_out;

  char* w = (char*)d_ws;
  auto alloc = [&](size_t bytes) -> void* {
    void* p = (void*)w;
    w += (bytes + 255) & ~(size_t)255;
    return p;
  };
  u16*   xbf    = (u16*)alloc((size_t)kBS * kH * 2);
  u16*   wqT    = (u16*)alloc((size_t)kNH * kH * kD * 2);
  u16*   wkT    = (u16*)alloc((size_t)kNH * kH * kD * 2);
  u16*   wvT    = (u16*)alloc((size_t)kNH * kH * kD * 2);
  u16*   woT    = (u16*)alloc((size_t)kH * kH * 2);
  u16*   w1T    = (u16*)alloc((size_t)kH * kFF * 2);
  u16*   w2T    = (u16*)alloc((size_t)kFF * kH * 2);
  u16*   qb     = (u16*)alloc((size_t)kNH * kBS * kD * 2);
  u16*   kb     = (u16*)alloc((size_t)kNH * kBS * kD * 2);
  u16*   vtb    = (u16*)alloc((size_t)kNH * kBS * kD * 2);
  u16*   attnbf = (u16*)alloc((size_t)kBS * kH * 2);
  float* projf  = (float*)alloc((size_t)kBS * kH * 4);
  float* n1f    = (float*)alloc((size_t)kBS * kH * 4);
  u16*   n1bf   = (u16*)alloc((size_t)kBS * kH * 2);
  u16*   ff1bf  = (u16*)alloc((size_t)kBS * kFF * 2);
  float* ff2f   = (float*)alloc((size_t)kBS * kH * 4);
  if ((size_t)(w - (char*)d_ws) > ws_size) return;  // ws too small

  // 1) input -> bf16
  {
    long n = (long)kBS * kH;
    cvt_bf16_kernel<<<dim3((unsigned)((n + 255) / 256)), dim3(256), 0, stream>>>(x, xbf, n);
  }
  // 2) weight transposes (-> [N,K] bf16)
  transpose_bf16_kernel<<<dim3(192, 1, kNH), dim3(256), 0, stream>>>(Wq, wqT, kH, kD);
  transpose_bf16_kernel<<<dim3(192, 1, kNH), dim3(256), 0, stream>>>(Wk, wkT, kH, kD);
  transpose_bf16_kernel<<<dim3(192, 1, kNH), dim3(256), 0, stream>>>(Wv, wvT, kH, kD);
  transpose_bf16_kernel<<<dim3(2304, 1, 1), dim3(256), 0, stream>>>(Wo, woT, kH, kH);
  transpose_bf16_kernel<<<dim3(9216, 1, 1), dim3(256), 0, stream>>>(W1, w1T, kH, kFF);
  transpose_bf16_kernel<<<dim3(9216, 1, 1), dim3(256), 0, stream>>>(W2, w2T, kFF, kH);

  // 3) Q/K/V projections (per head via z)
  const long sBt = (long)kH * kD;
  const long sQ  = (long)kBS * kD;
  gemm_bf16_wmma<<<dim3(1, 64, kNH), dim3(128), 0, stream>>>(
      xbf, wqT, nullptr, nullptr, qb, kBS, kD, kH, 0, 0, sBt, 0, sQ);
  gemm_bf16_wmma<<<dim3(1, 64, kNH), dim3(128), 0, stream>>>(
      xbf, wkT, nullptr, nullptr, kb, kBS, kD, kH, 0, 0, sBt, 0, sQ);
  gemm_bf16_wmma<<<dim3(1, 64, kNH), dim3(128), 0, stream>>>(
      xbf, wvT, nullptr, nullptr, vtb, kBS, kD, kH, 0, 1, sBt, 0, sQ);

  // 4) flash attention -> concat bf16 [4096,768]
  flash_attn_kernel<<<dim3(128, kNH, 2), dim3(32), 0, stream>>>(qb, kb, vtb, attnbf);

  // 5) output projection + bias -> f32
  gemm_bf16_wmma<<<dim3(12, 64, 1), dim3(128), 0, stream>>>(
      attnbf, woT, bo, projf, nullptr, kBS, kH, kH, 0, 0, 0, 0, 0);

  // 6) residual + LN1 -> f32 + bf16
  add_ln_kernel<<<dim3(kBS), dim3(256), 0, stream>>>(projf, x, g1, be1, n1f, n1bf);

  // 7) FFN: relu(n1 @ W1 + b1) -> bf16
  gemm_bf16_wmma<<<dim3(48, 64, 1), dim3(128), 0, stream>>>(
      n1bf, w1T, b1, nullptr, ff1bf, kBS, kFF, kH, 1, 0, 0, 0, 0);

  // 8) FFN: ff1 @ W2 + b2 -> f32
  gemm_bf16_wmma<<<dim3(12, 64, 1), dim3(128), 0, stream>>>(
      ff1bf, w2T, b2, ff2f, nullptr, kBS, kH, kFF, 0, 0, 0, 0, 0);

  // 9) residual + LN2 -> d_out (f32)
  add_ln_kernel<<<dim3(kBS), dim3(256), 0, stream>>>(ff2f, n1f, g2, be2, outp, nullptr);
}
